// LSTM_51453708206505
// MI455X (gfx1250) — matrix operand order repaired and checked by reading, not compiled, not run on hardware
//
#include <hip/hip_runtime.h>
#include <math.h>

// ---------------------------------------------------------------------------
// 2-layer LSTM (B=64,T=512,I=H=1024) + FC(400) for MI455X (gfx1250, wave32).
// All GEMMs run on v_wmma_f32_16x16x32_bf16. bf16 weights (~33MB) stay L2-
// resident (192MB L2), so the sequential recurrence is matrix-core bound.
// ---------------------------------------------------------------------------

typedef __attribute__((ext_vector_type(16))) __bf16 v16bf;
typedef __attribute__((ext_vector_type(8)))  __bf16 v8bf;
typedef __attribute__((ext_vector_type(8)))  float  v8f;

#define B_ 64
#define T_ 512
#define I_ 1024
#define H_ 1024
#define C_ 400

__device__ __forceinline__ unsigned short f32_to_bf16_rne(float f) {
  unsigned int u = __float_as_uint(f);
  u += 0x7FFFu + ((u >> 16) & 1u);      // round to nearest even
  return (unsigned short)(u >> 16);
}

__global__ void cvt_f32_bf16(const float* __restrict__ in,
                             unsigned short* __restrict__ out, long n) {
  long i = (long)blockIdx.x * blockDim.x + threadIdx.x;
  long stride = (long)gridDim.x * blockDim.x;
  for (; i < n; i += stride) out[i] = f32_to_bf16_rne(in[i]);
}

__global__ void zero_u32(unsigned int* __restrict__ p, long n) {
  long i = (long)blockIdx.x * blockDim.x + threadIdx.x;
  long stride = (long)gridDim.x * blockDim.x;
  for (; i < n; i += stride) p[i] = 0u;
}

// Load 16 consecutive bf16 (B fragment half-lane layout: K = kbase..kbase+15).
__device__ __forceinline__ v16bf load_bf16x16(const unsigned short* p) {
  v8bf lo = *(const v8bf*)p;
  v8bf hi = *(const v8bf*)(p + 8);
  return __builtin_shufflevector(lo, hi, 0,1,2,3,4,5,6,7,8,9,10,11,12,13,14,15);
}

__device__ __forceinline__ float sigmoidf_dev(float x) {
  return 1.0f / (1.0f + __expf(-x));
}

// One LSTM timestep for one layer.
// z = x_t @ Wih^T + h_prev @ Whh^T + bih + bhh   (z: [64, 4096])
// Each wave32 block owns a 16x16 (m,n) tile within H and computes it for all
// 4 gates (row offsets 0,H,2H,3H of W), then does the fused pointwise cell.
__global__ __launch_bounds__(32) void lstm_step_kernel(
    const unsigned short* __restrict__ xt,   long lda_x, // input slice, row stride
    const unsigned short* __restrict__ hprev,            // [64,H] bf16, stride H
    const unsigned short* __restrict__ Wih,              // [4H, K=1024] bf16
    const unsigned short* __restrict__ Whh,              // [4H, H=1024] bf16
    const float* __restrict__ bih, const float* __restrict__ bhh, // [4H]
    float* __restrict__ cstate,                          // [64,H] f32 in/out
    unsigned short* __restrict__ hnext,                  // [64,H] bf16 out
    float* __restrict__ hout_f32)                        // optional [64,H] f32 out
{
  const int lane  = threadIdx.x;
  const int nt    = blockIdx.x;          // N tile within H: 0..63
  const int mt    = blockIdx.y;          // M tile: 0..3
  const int nl    = lane & 15;
  const int half  = lane >> 4;
  const int m     = mt * 16 + nl;        // A-fragment row for this lane
  const int koff  = half * 16;           // A: K split matching B layout (contiguous 16 per half)
  const int kbase = half * 16;           // B: K split matching A (contiguous 16 per half)
  const int nglob = nt * 16 + nl;        // output column within H

  v8f acc0 = {}, acc1 = {}, acc2 = {}, acc3 = {};

  #pragma unroll 1
  for (int src = 0; src < 2; ++src) {
    const unsigned short* arow =
        (src == 0) ? (xt + (long)m * lda_x) : (hprev + (long)m * H_);
    const unsigned short* w = (src == 0) ? Wih : Whh;
    const unsigned short* w0 = w + ((long)(0 * H_) + nglob) * 1024;
    const unsigned short* w1 = w + ((long)(1 * H_) + nglob) * 1024;
    const unsigned short* w2 = w + ((long)(2 * H_) + nglob) * 1024;
    const unsigned short* w3 = w + ((long)(3 * H_) + nglob) * 1024;
    #pragma unroll 1
    for (int k0 = 0; k0 < 1024; k0 += 32) {
      // A fragment: row m, elements 0..7 -> K=k0+koff.., 8..15 -> K=k0+8+koff..
      v8bf alo = *(const v8bf*)(arow + k0 + koff);
      v8bf ahi = *(const v8bf*)(arow + k0 + 8 + koff);
      v16bf a = __builtin_shufflevector(alo, ahi,
                 0,1,2,3,4,5,6,7,8,9,10,11,12,13,14,15);
      v16bf b0 = load_bf16x16(w0 + k0 + kbase);
      acc0 = __builtin_amdgcn_wmma_f32_16x16x32_bf16(false, a, false, b0,
                                                     (short)0, acc0, false, false);
      v16bf b1 = load_bf16x16(w1 + k0 + kbase);
      acc1 = __builtin_amdgcn_wmma_f32_16x16x32_bf16(false, a, false, b1,
                                                     (short)0, acc1, false, false);
      v16bf b2 = load_bf16x16(w2 + k0 + kbase);
      acc2 = __builtin_amdgcn_wmma_f32_16x16x32_bf16(false, a, false, b2,
                                                     (short)0, acc2, false, false);
      v16bf b3 = load_bf16x16(w3 + k0 + kbase);
      acc3 = __builtin_amdgcn_wmma_f32_16x16x32_bf16(false, a, false, b3,
                                                     (short)0, acc3, false, false);
    }
  }

  // Fused LSTM cell. C/D layout: VGPR r, lane l -> M = r + 8*(l>>4), N = l&15.
  const float bi0 = bih[0 * H_ + nglob] + bhh[0 * H_ + nglob];
  const float bi1 = bih[1 * H_ + nglob] + bhh[1 * H_ + nglob];
  const float bi2 = bih[2 * H_ + nglob] + bhh[2 * H_ + nglob];
  const float bi3 = bih[3 * H_ + nglob] + bhh[3 * H_ + nglob];
  #pragma unroll
  for (int r = 0; r < 8; ++r) {
    const int  mr  = mt * 16 + half * 8 + r;
    const long idx = (long)mr * H_ + nglob;
    const float ig = sigmoidf_dev(acc0[r] + bi0);
    const float fg = sigmoidf_dev(acc1[r] + bi1);
    const float gg = tanhf(acc2[r] + bi2);
    const float og = sigmoidf_dev(acc3[r] + bi3);
    const float cn = fg * cstate[idx] + ig * gg;
    const float hn = og * tanhf(cn);
    cstate[idx] = cn;
    hnext[idx]  = f32_to_bf16_rne(hn);
    if (hout_f32) hout_f32[idx] = hn;
  }
}

// logits[64,400] = feat[64,1024] @ fcW^T + fcb    (400 = 25 * 16 exactly)
__global__ __launch_bounds__(32) void fc_kernel(
    const unsigned short* __restrict__ feat,  // [64,1024] bf16
    const unsigned short* __restrict__ W,     // [400,1024] bf16
    const float* __restrict__ bias,           // [400]
    float* __restrict__ out)                  // [64,400] f32
{
  const int lane  = threadIdx.x;
  const int nt    = blockIdx.x;          // 0..24
  const int mt    = blockIdx.y;          // 0..3
  const int nl    = lane & 15;
  const int half  = lane >> 4;
  const int m     = mt * 16 + nl;
  const int koff  = half * 16;
  const int kbase = half * 16;
  const int nglob = nt * 16 + nl;        // < 400

  v8f acc = {};
  const unsigned short* arow = feat + (long)m * 1024;
  const unsigned short* wrow = W + (long)nglob * 1024;
  #pragma unroll 1
  for (int k0 = 0; k0 < 1024; k0 += 32) {
    v8bf alo = *(const v8bf*)(arow + k0 + koff);
    v8bf ahi = *(const v8bf*)(arow + k0 + 8 + koff);
    v16bf a = __builtin_shufflevector(alo, ahi,
               0,1,2,3,4,5,6,7,8,9,10,11,12,13,14,15);
    v16bf b = load_bf16x16(wrow + k0 + kbase);
    acc = __builtin_amdgcn_wmma_f32_16x16x32_bf16(false, a, false, b,
                                                  (short)0, acc, false, false);
  }
  const float bv = bias[nglob];
  #pragma unroll
  for (int r = 0; r < 8; ++r) {
    const int mr = mt * 16 + half * 8 + r;
    out[(long)mr * C_ + nglob] = acc[r] + bv;
  }
}

extern "C" void kernel_launch(void* const* d_in, const int* in_sizes, int n_in,
                              void* d_out, int out_size, void* d_ws, size_t ws_size,
                              hipStream_t stream) {
  const float* x    = (const float*)d_in[0];
  const float* Wih0 = (const float*)d_in[1];
  const float* Whh0 = (const float*)d_in[2];
  const float* bih0 = (const float*)d_in[3];
  const float* bhh0 = (const float*)d_in[4];
  const float* Wih1 = (const float*)d_in[5];
  const float* Whh1 = (const float*)d_in[6];
  const float* bih1 = (const float*)d_in[7];
  const float* bhh1 = (const float*)d_in[8];
  const float* fcW  = (const float*)d_in[9];
  const float* fcb  = (const float*)d_in[10];
  float* out = (float*)d_out;                 // [64*400 logits][64*1024 feat]

  // ---- workspace carve-out (~170 MB) ----
  char* ws = (char*)d_ws;
  size_t off = 0;
  auto carve = [&](size_t bytes) -> char* {
    char* p = ws + off;
    off += (bytes + 255) & ~(size_t)255;
    return p;
  };
  unsigned short* x_bf   = (unsigned short*)carve((size_t)B_ * T_ * I_ * 2);
  unsigned short* h1_all = (unsigned short*)carve((size_t)T_ * B_ * H_ * 2); // [t][b][h]
  unsigned short* Wih0b  = (unsigned short*)carve((size_t)4 * H_ * I_ * 2);
  unsigned short* Whh0b  = (unsigned short*)carve((size_t)4 * H_ * H_ * 2);
  unsigned short* Wih1b  = (unsigned short*)carve((size_t)4 * H_ * H_ * 2);
  unsigned short* Whh1b  = (unsigned short*)carve((size_t)4 * H_ * H_ * 2);
  unsigned short* fcWb   = (unsigned short*)carve((size_t)C_ * H_ * 2);
  float*          cstate = (float*)         carve((size_t)B_ * H_ * 4);
  unsigned short* hzero  = (unsigned short*)carve((size_t)B_ * H_ * 2);
  unsigned short* h2a    = (unsigned short*)carve((size_t)B_ * H_ * 2);
  unsigned short* h2b    = (unsigned short*)carve((size_t)B_ * H_ * 2);
  (void)in_sizes; (void)n_in; (void)out_size; (void)ws_size;

  // ---- one-time-per-call conversions (deterministic, inputs untouched) ----
  cvt_f32_bf16<<<2048, 256, 0, stream>>>(x,    x_bf,  (long)B_ * T_ * I_);
  cvt_f32_bf16<<<1024, 256, 0, stream>>>(Wih0, Wih0b, (long)4 * H_ * I_);
  cvt_f32_bf16<<<1024, 256, 0, stream>>>(Whh0, Whh0b, (long)4 * H_ * H_);
  cvt_f32_bf16<<<1024, 256, 0, stream>>>(Wih1, Wih1b, (long)4 * H_ * H_);
  cvt_f32_bf16<<<1024, 256, 0, stream>>>(fcW,  fcWb,  (long)C_ * H_);
  cvt_f32_bf16<<<1024, 256, 0, stream>>>(Whh1, Whh1b, (long)4 * H_ * H_);
  zero_u32<<<64, 256, 0, stream>>>((unsigned int*)cstate, (long)B_ * H_);
  zero_u32<<<64, 256, 0, stream>>>((unsigned int*)hzero,  (long)B_ * H_ / 2);

  const dim3 grid(H_ / 16, B_ / 16), blk(32);

  // ---- layer 0: x is [B,T,I] so step-t rows have stride T*I ----
  for (int t = 0; t < T_; ++t) {
    const unsigned short* xt = x_bf + (size_t)t * I_;
    const unsigned short* hp =
        (t == 0) ? hzero : (h1_all + (size_t)(t - 1) * B_ * H_);
    unsigned short* hn = h1_all + (size_t)t * B_ * H_;
    lstm_step_kernel<<<grid, blk, 0, stream>>>(
        xt, (long)T_ * I_, hp, Wih0b, Whh0b, bih0, bhh0, cstate, hn, nullptr);
  }

  // ---- layer 1: input = h1_all[t] (contiguous, stride H); ping-pong h ----
  zero_u32<<<64, 256, 0, stream>>>((unsigned int*)cstate, (long)B_ * H_);
  float* feat_out = out + (size_t)B_ * C_;   // fp32 feat region of d_out
  for (int t = 0; t < T_; ++t) {
    const unsigned short* xt = h1_all + (size_t)t * B_ * H_;
    const unsigned short* hp = (t == 0) ? hzero : ((t & 1) ? h2a : h2b);
    unsigned short* hn = (t & 1) ? h2b : h2a;
    lstm_step_kernel<<<grid, blk, 0, stream>>>(
        xt, (long)H_, hp, Wih1b, Whh1b, bih1, bhh1, cstate, hn,
        (t == T_ - 1) ? feat_out : nullptr);
  }
  const unsigned short* feat_bf = h2b;       // T-1 = 511 is odd -> wrote h2b

  // ---- final FC: logits into d_out[0 .. 64*400) ----
  fc_kernel<<<dim3(C_ / 16, B_ / 16), blk, 0, stream>>>(feat_bf, fcWb, fcb, out);
}
